// CharAutoencoder_38250978738250
// MI455X (gfx1250) — compile-verified, matrix operand order checked by reading, not run on hardware
//
#include <hip/hip_runtime.h>

typedef __bf16 bf16;
typedef __attribute__((ext_vector_type(16))) __bf16 v16bf;
typedef __attribute__((ext_vector_type(8)))  float  v8f;

#define BATCH   1024
#define SEQL    128
#define EMBD    64
#define HID     128
#define GATES   512
#define VOCABSZ 128

__device__ __forceinline__ float sigf(float x) { return 1.0f / (1.0f + __expf(-x)); }

// ---------------------------------------------------------------------------
// A-matrix fragment (16x32 bf16, M x K) built from an LDS tile [16 x rs].
// Layout per ISA 7.12.2: lane m=lane&15; lanes>=16 add K+8; VGPR r<4: K=2r,2r+1;
// VGPR r>=4: K=16+2(r-4)+... -> k = (r>>2)*16 + (r&3)*2 (+8 hi half).
// ---------------------------------------------------------------------------
__device__ __forceinline__ v16bf load_a_frag(const bf16* base, int rs, int kc, int lane) {
  union { unsigned u[8]; v16bf v; } a;
  const bf16* row = base + (lane & 15) * rs + kc * 32 + ((lane >> 4) << 3);
#pragma unroll
  for (int r = 0; r < 8; ++r) {
    int k = ((r >> 2) << 4) + ((r & 3) << 1);
    a.u[r] = *reinterpret_cast<const unsigned*>(row + k);
  }
  return a.v;
}

// B fragments are pre-swizzled: fragment fidx, lane-> 16 contiguous bf16 (32B).
__device__ __forceinline__ v16bf load_b_frag(const bf16* frags, int fidx, int lane) {
  return *reinterpret_cast<const v16bf*>(frags + (((size_t)fidx << 5) + lane) * 16);
}

// ---------------------------------------------------------------------------
// Pre-swizzle weight W[N,K] (row-major, gate-major) into B-fragment layout:
// frag f = ntile*(K/32)+kc ; B[k][n] = W[ntile*16 + (lane&15)][k],
// k = kc*32 + (e>>3)*16 + ((e>>1)&3)*2 + (e&1) + (lane>=16 ? 8 : 0)
// ---------------------------------------------------------------------------
__global__ void prep_b_frags(const float* __restrict__ W, bf16* __restrict__ out,
                             int N, int K) {
  int gid = blockIdx.x * blockDim.x + threadIdx.x;
  int nfrag = (N / 16) * (K / 32);
  if (gid >= nfrag * 32) return;
  int f = gid >> 5, lane = gid & 31;
  int KC = K / 32;
  int kc = f % KC, nt = f / KC;
  int n = nt * 16 + (lane & 15);
  int hi = (lane >> 4) * 8;
  bf16* dst = out + (size_t)gid * 16;
#pragma unroll
  for (int e = 0; e < 16; ++e) {
    int k = kc * 32 + ((e >> 3) << 4) + (((e >> 1) & 3) << 1) + (e & 1) + hi;
    dst[e] = (bf16)W[(size_t)n * K + k];
  }
}

__global__ void prep_bias(const float* __restrict__ a, const float* __restrict__ b,
                          float* __restrict__ out) {
  int i = blockIdx.x * blockDim.x + threadIdx.x;
  if (i < GATES) out[i] = a[i] + b[i];
}

// Embedding gather -> bf16 [B, L, 64]. shifted=1: decoder teacher forcing input.
__global__ void embed_kernel(const int* __restrict__ toks, const float* __restrict__ embed,
                             bf16* __restrict__ out, int shifted) {
  int bt = blockIdx.x * blockDim.x + threadIdx.x;
  if (bt >= BATCH * SEQL) return;
  int b = bt >> 7, t = bt & (SEQL - 1);
  int tok = shifted ? ((t == 0) ? 0 : toks[b * SEQL + t - 1]) : toks[bt];
  const float* src = embed + (size_t)tok * EMBD;
  bf16* dst = out + (size_t)bt * EMBD;
#pragma unroll
  for (int e = 0; e < EMBD; ++e) dst[e] = (bf16)src[e];
}

// ---------------------------------------------------------------------------
// Persistent bidirectional LSTM layer. 128 blocks: 0..63 forward, 64..127
// backward. Each block owns a 16-row batch tile and scans all 128 steps.
// 8 waves x 4 gate-tiles of 16 cover the 512 gates. Weights stream from L2
// as pre-swizzled bf16 B-fragments.
// ---------------------------------------------------------------------------
template <int IN_DIM, bool WRITE_H, bool POOL>
__global__ __launch_bounds__(256) void enc_lstm(
    const bf16* __restrict__ in,
    const bf16* __restrict__ wihF, const bf16* __restrict__ whhF, const float* __restrict__ biasF,
    const bf16* __restrict__ wihB, const bf16* __restrict__ whhB, const float* __restrict__ biasB,
    bf16* __restrict__ out_h, float* __restrict__ pooled) {
  __shared__ __attribute__((aligned(16))) bf16 xbf[16 * IN_DIM];
  __shared__ __attribute__((aligned(16))) bf16 hbf[16 * HID];
  __shared__ __attribute__((aligned(16))) float gbuf[16 * GATES];

  const int dir = blockIdx.x >> 6;
  const int b0 = (blockIdx.x & 63) << 4;
  const int tid = threadIdx.x;
  const int wave = tid >> 5, lane = tid & 31, m16 = lane & 31 & 15;
  const int flat = tid << 3;

  const bf16* wih = dir ? wihB : wihF;
  const bf16* whh = dir ? whhB : whhF;
  const float* bias = dir ? biasB : biasF;
  const int col_off = dir ? HID : 0;

  float c_reg[8], pool_acc[8];
#pragma unroll
  for (int q = 0; q < 8; ++q) {
    c_reg[q] = 0.f;
    pool_acc[q] = 0.f;
    hbf[flat + q] = (bf16)0.f;
  }
  float bias_v[4];
#pragma unroll
  for (int nt = 0; nt < 4; ++nt) bias_v[nt] = bias[wave * 64 + nt * 16 + m16];
  __syncthreads();

  const int KCX = IN_DIM / 32;
  for (int t = 0; t < SEQL; ++t) {
    const int tt = dir ? (SEQL - 1 - t) : t;
    // stage x_t tile [16 x IN_DIM] into LDS (dword-granular, coalesced)
    for (int u = tid; u < (16 * IN_DIM) / 2; u += 256) {
      int row = (u << 1) / IN_DIM, col = (u << 1) % IN_DIM;
      *reinterpret_cast<unsigned*>(xbf + row * IN_DIM + col) =
          *reinterpret_cast<const unsigned*>(in + ((size_t)(b0 + row) * SEQL + tt) * IN_DIM + col);
    }
    __syncthreads();

#pragma unroll
    for (int nt = 0; nt < 4; ++nt) {
      const int ntg = wave * 4 + nt;
      v8f acc;
#pragma unroll
      for (int r = 0; r < 8; ++r) acc[r] = bias_v[nt];
#pragma unroll
      for (int kc = 0; kc < KCX; ++kc) {
        v16bf a = load_a_frag(xbf, IN_DIM, kc, lane);
        v16bf b = load_b_frag(wih, ntg * KCX + kc, lane);
        acc = __builtin_amdgcn_wmma_f32_16x16x32_bf16(false, a, false, b, (short)0, acc, false, false);
      }
#pragma unroll
      for (int kc = 0; kc < 4; ++kc) {
        v16bf a = load_a_frag(hbf, HID, kc, lane);
        v16bf b = load_b_frag(whh, ntg * 4 + kc, lane);
        acc = __builtin_amdgcn_wmma_f32_16x16x32_bf16(false, a, false, b, (short)0, acc, false, false);
      }
      const int rowoff = (lane >> 4) << 3;  // C/D layout: lanes>=16 hold M=8..15
#pragma unroll
      for (int r = 0; r < 8; ++r)
        gbuf[(r + rowoff) * GATES + ntg * 16 + m16] = acc[r];
    }
    __syncthreads();

    // fused LSTM cell: each thread owns 8 (m,j) cells (PyTorch gate order i,f,g,o)
#pragma unroll
    for (int q = 0; q < 8; ++q) {
      const int e = flat + q, mm = e >> 7, jj = e & (HID - 1);
      float gi = gbuf[mm * GATES + jj];
      float gf = gbuf[mm * GATES + HID + jj];
      float gg = gbuf[mm * GATES + 2 * HID + jj];
      float go = gbuf[mm * GATES + 3 * HID + jj];
      float cc = sigf(gf) * c_reg[q] + sigf(gi) * tanhf(gg);
      c_reg[q] = cc;
      float hh = sigf(go) * tanhf(cc);
      hbf[mm * HID + jj] = (bf16)hh;
      if (WRITE_H)
        out_h[((size_t)(b0 + mm) * SEQL + tt) * 256 + col_off + jj] = (bf16)hh;
      if (POOL) pool_acc[q] += hh;
    }
    __syncthreads();
  }

  if (POOL) {
#pragma unroll
    for (int q = 0; q < 8; ++q) {
      const int e = flat + q, mm = e >> 7, jj = e & (HID - 1);
      pooled[(size_t)(b0 + mm) * 256 + col_off + jj] = pool_acc[q] * (1.0f / SEQL);
    }
  }
}

// ---------------------------------------------------------------------------
// Decoder LSTM with h0 from bottleneck + fused vocab projection into d_out.
// ---------------------------------------------------------------------------
__global__ __launch_bounds__(256) void dec_lstm(
    const bf16* __restrict__ in,
    const bf16* __restrict__ wih, const bf16* __restrict__ whh, const float* __restrict__ bias,
    const bf16* __restrict__ outw, const float* __restrict__ outb,
    const float* __restrict__ h0, float* __restrict__ logits) {
  __shared__ __attribute__((aligned(16))) bf16 xbf[16 * EMBD];
  __shared__ __attribute__((aligned(16))) bf16 hbf[16 * HID];
  __shared__ __attribute__((aligned(16))) float gbuf[16 * GATES];

  const int b0 = blockIdx.x << 4;
  const int tid = threadIdx.x;
  const int wave = tid >> 5, lane = tid & 31, m16 = lane & 15;
  const int flat = tid << 3;

  float c_reg[8];
#pragma unroll
  for (int q = 0; q < 8; ++q) {
    const int e = flat + q, mm = e >> 7, jj = e & (HID - 1);
    c_reg[q] = 0.f;
    hbf[e] = (bf16)h0[(size_t)(b0 + mm) * HID + jj];
  }
  float bias_v[4];
#pragma unroll
  for (int nt = 0; nt < 4; ++nt) bias_v[nt] = bias[wave * 64 + nt * 16 + m16];
  const float ob = outb[wave * 16 + m16];  // this wave's vocab tile
  __syncthreads();

  for (int t = 0; t < SEQL; ++t) {
    for (int u = tid; u < (16 * EMBD) / 2; u += 256) {
      int row = (u << 1) / EMBD, col = (u << 1) % EMBD;
      *reinterpret_cast<unsigned*>(xbf + row * EMBD + col) =
          *reinterpret_cast<const unsigned*>(in + ((size_t)(b0 + row) * SEQL + t) * EMBD + col);
    }
    __syncthreads();

#pragma unroll
    for (int nt = 0; nt < 4; ++nt) {
      const int ntg = wave * 4 + nt;
      v8f acc;
#pragma unroll
      for (int r = 0; r < 8; ++r) acc[r] = bias_v[nt];
#pragma unroll
      for (int kc = 0; kc < 2; ++kc) {  // K = 64
        v16bf a = load_a_frag(xbf, EMBD, kc, lane);
        v16bf b = load_b_frag(wih, ntg * 2 + kc, lane);
        acc = __builtin_amdgcn_wmma_f32_16x16x32_bf16(false, a, false, b, (short)0, acc, false, false);
      }
#pragma unroll
      for (int kc = 0; kc < 4; ++kc) {  // K = 128
        v16bf a = load_a_frag(hbf, HID, kc, lane);
        v16bf b = load_b_frag(whh, ntg * 4 + kc, lane);
        acc = __builtin_amdgcn_wmma_f32_16x16x32_bf16(false, a, false, b, (short)0, acc, false, false);
      }
      const int rowoff = (lane >> 4) << 3;
#pragma unroll
      for (int r = 0; r < 8; ++r)
        gbuf[(r + rowoff) * GATES + ntg * 16 + m16] = acc[r];
    }
    __syncthreads();

#pragma unroll
    for (int q = 0; q < 8; ++q) {
      const int e = flat + q, mm = e >> 7, jj = e & (HID - 1);
      float gi = gbuf[mm * GATES + jj];
      float gf = gbuf[mm * GATES + HID + jj];
      float gg = gbuf[mm * GATES + 2 * HID + jj];
      float go = gbuf[mm * GATES + 3 * HID + jj];
      float cc = sigf(gf) * c_reg[q] + sigf(gi) * tanhf(gg);
      c_reg[q] = cc;
      float hh = sigf(go) * tanhf(cc);
      hbf[mm * HID + jj] = (bf16)hh;
    }
    __syncthreads();

    // fused projection: logits[16 x 128] = h @ out_W^T + out_b ; wave = vocab tile
    {
      v8f acc;
#pragma unroll
      for (int r = 0; r < 8; ++r) acc[r] = ob;
#pragma unroll
      for (int kc = 0; kc < 4; ++kc) {
        v16bf a = load_a_frag(hbf, HID, kc, lane);
        v16bf b = load_b_frag(outw, wave * 4 + kc, lane);
        acc = __builtin_amdgcn_wmma_f32_16x16x32_bf16(false, a, false, b, (short)0, acc, false, false);
      }
      const int rowoff = (lane >> 4) << 3;
#pragma unroll
      for (int r = 0; r < 8; ++r)
        logits[((size_t)(b0 + r + rowoff) * SEQL + t) * VOCABSZ + wave * 16 + m16] = acc[r];
    }
  }
}

// ---------------------------------------------------------------------------
// Bottleneck MLP 256->64->16->6 + LayerNorm(6) + 6->128. One thread per row.
// ---------------------------------------------------------------------------
__global__ void bottleneck_kernel(const float* __restrict__ pooled,
                                  const float* __restrict__ W1, const float* __restrict__ b1,
                                  const float* __restrict__ W2, const float* __restrict__ b2,
                                  const float* __restrict__ W3, const float* __restrict__ b3,
                                  const float* __restrict__ lng, const float* __restrict__ lnb,
                                  const float* __restrict__ b2dW, const float* __restrict__ b2db,
                                  float* __restrict__ hidden) {
  int b = blockIdx.x * blockDim.x + threadIdx.x;
  if (b >= BATCH) return;
  const float* x = pooled + (size_t)b * 256;
  float v1[64];
  for (int o = 0; o < 64; ++o) {
    float s = b1[o];
    for (int k = 0; k < 256; ++k) s += W1[o * 256 + k] * x[k];
    v1[o] = fmaxf(s, 0.f);
  }
  float v2[16];
  for (int o = 0; o < 16; ++o) {
    float s = b2[o];
    for (int k = 0; k < 64; ++k) s += W2[o * 64 + k] * v1[k];
    v2[o] = fmaxf(s, 0.f);
  }
  float v3[6];
  float mu = 0.f;
  for (int o = 0; o < 6; ++o) {
    float s = b3[o];
    for (int k = 0; k < 16; ++k) s += W3[o * 16 + k] * v2[k];
    v3[o] = s;
    mu += s;
  }
  mu *= (1.f / 6.f);
  float var = 0.f;
  for (int o = 0; o < 6; ++o) { float d = v3[o] - mu; var += d * d; }
  var *= (1.f / 6.f);
  float rs = rsqrtf(var + 1e-5f);
  for (int o = 0; o < 6; ++o) v3[o] = (v3[o] - mu) * rs * lng[o] + lnb[o];
  float* hd = hidden + (size_t)b * HID;
  for (int o = 0; o < HID; ++o) {
    float s = b2db[o];
    for (int k = 0; k < 6; ++k) s += b2dW[o * 6 + k] * v3[k];
    hd[o] = s;
  }
}

// ---------------------------------------------------------------------------
extern "C" void kernel_launch(void* const* d_in, const int* in_sizes, int n_in,
                              void* d_out, int out_size, void* d_ws, size_t ws_size,
                              hipStream_t stream) {
  (void)in_sizes; (void)n_in; (void)out_size; (void)ws_size;
  // setup_inputs order: x, targets, embed, {enc0f,enc0b,enc1f,enc1b}x{Wih,Whh,bih,bhh},
  // bn_W1,bn_b1,bn_W2,bn_b2,bn_W3,bn_b3, ln_g,ln_b, b2d_W,b2d_b,
  // dec_Wih,dec_Whh,dec_bih,dec_bhh, out_W,out_b
  const int* x = (const int*)d_in[0];
  const int* targets = (const int*)d_in[1];
  const float* embed = (const float*)d_in[2];
  const float* e0f_Wih = (const float*)d_in[3];
  const float* e0f_Whh = (const float*)d_in[4];
  const float* e0f_bih = (const float*)d_in[5];
  const float* e0f_bhh = (const float*)d_in[6];
  const float* e0b_Wih = (const float*)d_in[7];
  const float* e0b_Whh = (const float*)d_in[8];
  const float* e0b_bih = (const float*)d_in[9];
  const float* e0b_bhh = (const float*)d_in[10];
  const float* e1f_Wih = (const float*)d_in[11];
  const float* e1f_Whh = (const float*)d_in[12];
  const float* e1f_bih = (const float*)d_in[13];
  const float* e1f_bhh = (const float*)d_in[14];
  const float* e1b_Wih = (const float*)d_in[15];
  const float* e1b_Whh = (const float*)d_in[16];
  const float* e1b_bih = (const float*)d_in[17];
  const float* e1b_bhh = (const float*)d_in[18];
  const float* bnW1 = (const float*)d_in[19];
  const float* bnb1 = (const float*)d_in[20];
  const float* bnW2 = (const float*)d_in[21];
  const float* bnb2 = (const float*)d_in[22];
  const float* bnW3 = (const float*)d_in[23];
  const float* bnb3 = (const float*)d_in[24];
  const float* lng = (const float*)d_in[25];
  const float* lnb = (const float*)d_in[26];
  const float* b2dW = (const float*)d_in[27];
  const float* b2db = (const float*)d_in[28];
  const float* dec_Wih = (const float*)d_in[29];
  const float* dec_Whh = (const float*)d_in[30];
  const float* dec_bih = (const float*)d_in[31];
  const float* dec_bhh = (const float*)d_in[32];
  const float* out_W = (const float*)d_in[33];
  const float* out_b = (const float*)d_in[34];

  char* ws = (char*)d_ws;
  size_t off = 0;
  auto take = [&](size_t bytes) -> void* {
    void* p = ws + off;
    off += (bytes + 255) & ~(size_t)255;
    return p;
  };
  bf16* enc_emb = (bf16*)take((size_t)BATCH * SEQL * EMBD * 2);
  bf16* dec_emb = (bf16*)take((size_t)BATCH * SEQL * EMBD * 2);
  bf16* h01 = (bf16*)take((size_t)BATCH * SEQL * 256 * 2);
  float* pooled = (float*)take((size_t)BATCH * 256 * 4);
  float* hidden = (float*)take((size_t)BATCH * HID * 4);
  bf16* f_e0f_ih = (bf16*)take((size_t)GATES * 64 * 2);
  bf16* f_e0f_hh = (bf16*)take((size_t)GATES * 128 * 2);
  bf16* f_e0b_ih = (bf16*)take((size_t)GATES * 64 * 2);
  bf16* f_e0b_hh = (bf16*)take((size_t)GATES * 128 * 2);
  bf16* f_e1f_ih = (bf16*)take((size_t)GATES * 256 * 2);
  bf16* f_e1f_hh = (bf16*)take((size_t)GATES * 128 * 2);
  bf16* f_e1b_ih = (bf16*)take((size_t)GATES * 256 * 2);
  bf16* f_e1b_hh = (bf16*)take((size_t)GATES * 128 * 2);
  bf16* f_dec_ih = (bf16*)take((size_t)GATES * 64 * 2);
  bf16* f_dec_hh = (bf16*)take((size_t)GATES * 128 * 2);
  bf16* f_outw = (bf16*)take((size_t)VOCABSZ * HID * 2);
  float* bs_e0f = (float*)take(GATES * 4);
  float* bs_e0b = (float*)take(GATES * 4);
  float* bs_e1f = (float*)take(GATES * 4);
  float* bs_e1b = (float*)take(GATES * 4);
  float* bs_dec = (float*)take(GATES * 4);

  auto prep = [&](const float* W, bf16* dst, int N, int K) {
    int threads = (N / 16) * (K / 32) * 32;
    prep_b_frags<<<(threads + 255) / 256, 256, 0, stream>>>(W, dst, N, K);
  };
  prep(e0f_Wih, f_e0f_ih, GATES, 64);
  prep(e0f_Whh, f_e0f_hh, GATES, 128);
  prep(e0b_Wih, f_e0b_ih, GATES, 64);
  prep(e0b_Whh, f_e0b_hh, GATES, 128);
  prep(e1f_Wih, f_e1f_ih, GATES, 256);
  prep(e1f_Whh, f_e1f_hh, GATES, 128);
  prep(e1b_Wih, f_e1b_ih, GATES, 256);
  prep(e1b_Whh, f_e1b_hh, GATES, 128);
  prep(dec_Wih, f_dec_ih, GATES, 64);
  prep(dec_Whh, f_dec_hh, GATES, 128);
  prep(out_W, f_outw, VOCABSZ, HID);

  prep_bias<<<2, 256, 0, stream>>>(e0f_bih, e0f_bhh, bs_e0f);
  prep_bias<<<2, 256, 0, stream>>>(e0b_bih, e0b_bhh, bs_e0b);
  prep_bias<<<2, 256, 0, stream>>>(e1f_bih, e1f_bhh, bs_e1f);
  prep_bias<<<2, 256, 0, stream>>>(e1b_bih, e1b_bhh, bs_e1b);
  prep_bias<<<2, 256, 0, stream>>>(dec_bih, dec_bhh, bs_dec);

  embed_kernel<<<(BATCH * SEQL) / 256, 256, 0, stream>>>(x, embed, enc_emb, 0);
  embed_kernel<<<(BATCH * SEQL) / 256, 256, 0, stream>>>(targets, embed, dec_emb, 1);

  // encoder layer 0 (writes h01), layer 1 (fused mean-pool only)
  enc_lstm<64, true, false><<<128, 256, 0, stream>>>(
      enc_emb, f_e0f_ih, f_e0f_hh, bs_e0f, f_e0b_ih, f_e0b_hh, bs_e0b, h01, nullptr);
  enc_lstm<256, false, true><<<128, 256, 0, stream>>>(
      h01, f_e1f_ih, f_e1f_hh, bs_e1f, f_e1b_ih, f_e1b_hh, bs_e1b, nullptr, pooled);

  bottleneck_kernel<<<4, 256, 0, stream>>>(pooled, bnW1, bnb1, bnW2, bnb2, bnW3, bnb3,
                                           lng, lnb, b2dW, b2db, hidden);

  dec_lstm<<<64, 256, 0, stream>>>(dec_emb, f_dec_ih, f_dec_hh, bs_dec, f_outw, out_b,
                                   hidden, (float*)d_out);
}
